// DynamicMultiHeadAttention_61933428408320
// MI455X (gfx1250) — matrix operand order repaired and checked by reading, not compiled, not run on hardware
//
#include <hip/hip_runtime.h>
#include <stdint.h>

typedef __attribute__((ext_vector_type(16))) __bf16 v16bf;
typedef __attribute__((ext_vector_type(8)))  __bf16 v8bf;
typedef __attribute__((ext_vector_type(8)))  float  v8f;

#define D_MODEL 512
#define T_SEQ   1024
#define NTOK    4096   // B*T
#define NHEADS  8

__device__ __forceinline__ v8bf ld8(const __bf16* p) { return *(const v8bf*)p; }

__device__ __forceinline__ v16bf mkfrag(v8bf lo, v8bf hi) {
  v16bf r;
#pragma unroll
  for (int i = 0; i < 8; ++i) { r[i] = lo[i]; r[i + 8] = hi[i]; }
  return r;
}

__device__ __forceinline__ v8f wmma_bf16(v16bf a, v16bf b, v8f c) {
  return __builtin_amdgcn_wmma_f32_16x16x32_bf16(false, a, false, b, (short)0, c,
                                                 false, false);
}

// ---------------- head params: mq2[h][d] = m^2 * invscale_h, mv[h][d] = m ---
__global__ void k_prep(const float* __restrict__ hw, float* __restrict__ mq2,
                       float* __restrict__ mv) {
  __shared__ float starts_s[NHEADS], dims_s[NHEADS], invs_s[NHEADS];
  if (threadIdx.x == 0) {
    float w[NHEADS], e[NHEADS], mx = -3.0e38f, s = 0.f;
    for (int h = 0; h < NHEADS; ++h) { w[h] = hw[h]; mx = fmaxf(mx, w[h]); }
    for (int h = 0; h < NHEADS; ++h) { e[h] = __expf(w[h] - mx); s += e[h]; }
    float run = 0.f;
    for (int h = 0; h < NHEADS; ++h) {
      float g = e[h] / s;
      float d = fmaxf(0.f, 16.f + g * (float)(D_MODEL - 16 * NHEADS));
      dims_s[h] = d; starts_s[h] = run; run += d;
      invs_s[h] = rsqrtf(d + 1e-6f);
    }
  }
  __syncthreads();
  int d = threadIdx.x;  // 512 threads
  float fd = (float)d;
  for (int h = 0; h < NHEADS; ++h) {
    float st = starts_s[h], dim = dims_s[h];
    float L = 1.f / (1.f + __expf(-(fd - st) * 10.f));
    float R = 1.f / (1.f + __expf(-((st + dim) - fd) * 10.f));
    float m = L * R;
    mq2[h * D_MODEL + d] = m * m * invs_s[h];
    mv[h * D_MODEL + d]  = m;
  }
}

// ---------------- small utility kernels ------------------------------------
__global__ void k_cvt_bf16(const float* __restrict__ src, __bf16* __restrict__ dst, int n) {
  int i = blockIdx.x * blockDim.x + threadIdx.x;
  if (i < n) dst[i] = (__bf16)src[i];
}

__global__ void k_transpose_bf16(const float* __restrict__ src, __bf16* __restrict__ dst,
                                 int R, int C) {
  int i = blockIdx.x * blockDim.x + threadIdx.x;
  if (i < R * C) { int r = i / C, c = i % C; dst[c * R + r] = (__bf16)src[i]; }
}

__global__ void k_headsum(const __bf16* __restrict__ ctxh, __bf16* __restrict__ ctxbf, int n) {
  int i = blockIdx.x * blockDim.x + threadIdx.x;
  if (i < n) {
    float s = 0.f;
#pragma unroll
    for (int h = 0; h < NHEADS; ++h) s += (float)ctxh[(size_t)h * n + i];
    ctxbf[i] = (__bf16)s;
  }
}

// ---------------- generic bf16 WMMA GEMM, A[M,512] x Bt[N,512]^T ------------
// MODE 0: QKV epilogue (split cols to q / k / v-transposed), bf16 outputs
// MODE 1: out-proj epilogue: res = A@W + bias + residual (f32)
template <int MODE>
__global__ __launch_bounds__(256) void k_gemm(
    const __bf16* __restrict__ A, const __bf16* __restrict__ Bt,
    const float* __restrict__ bias,
    __bf16* __restrict__ q, __bf16* __restrict__ k, __bf16* __restrict__ vt,
    const float* __restrict__ resid, float* __restrict__ res) {
  const int K = 512;
  int tid = threadIdx.x, w = tid >> 5, lane = tid & 31;
  int lhalf = lane >> 4, l15 = lane & 15;
  int wm = w & 3, wn = w >> 2;
  int m0 = blockIdx.x * 64 + wm * 16;
  int n0 = blockIdx.y * 64 + wn * 32;
  v8f acc0 = {}, acc1 = {};
  const __bf16* Ap  = A  + (size_t)(m0 + l15) * K + 16 * lhalf;
  const __bf16* Bp0 = Bt + (size_t)(n0 + l15) * K + 16 * lhalf;
  const __bf16* Bp1 = Bp0 + (size_t)16 * K;
#pragma unroll 4
  for (int kk = 0; kk < K; kk += 32) {
    v16bf af = mkfrag(ld8(Ap + kk), ld8(Ap + kk + 8));
    v16bf b0 = mkfrag(ld8(Bp0 + kk), ld8(Bp0 + kk + 8));
    v16bf b1 = mkfrag(ld8(Bp1 + kk), ld8(Bp1 + kk + 8));
    acc0 = wmma_bf16(af, b0, acc0);
    acc1 = wmma_bf16(af, b1, acc1);
  }
#pragma unroll
  for (int nt = 0; nt < 2; ++nt) {
    v8f acc = nt ? acc1 : acc0;
#pragma unroll
    for (int r = 0; r < 8; ++r) {
      int Mg = m0 + r + 8 * lhalf;          // C layout: M = r + 8*(lane/16)
      int Ng = n0 + nt * 16 + l15;          //           N = lane%16
      float v = acc[r] + bias[Ng];
      if (MODE == 0) {
        int g = Ng >> 9, c = Ng & 511;
        if (g == 0)      q[(size_t)Mg * 512 + c] = (__bf16)v;
        else if (g == 1) k[(size_t)Mg * 512 + c] = (__bf16)v;
        else             vt[(size_t)c * NTOK + Mg] = (__bf16)v;  // V transposed
      } else {
        res[(size_t)Mg * 512 + Ng] = v + resid[(size_t)Mg * 512 + Ng];
      }
    }
  }
}

// ---------------- flash attention: one WG per (32-query tile, b, h) ---------
__global__ __launch_bounds__(256) void k_attn(
    const __bf16* __restrict__ qbf, const __bf16* __restrict__ kbf,
    const __bf16* __restrict__ vtbf,
    const float* __restrict__ mq2, const float* __restrict__ mv,
    const unsigned char* __restrict__ kpm, __bf16* __restrict__ ctxh) {
  __shared__ __align__(16) __bf16 Qs[32][520];   // masked Q tile (padded rows)
  __shared__ __align__(16) __bf16 Ps[32][72];    // probabilities (bf16)
  __shared__ float Ss[32][65];                   // raw scores
  __shared__ float row_m[32], row_l[32], row_al[32], red[32][8];

  int qt = blockIdx.x, b = blockIdx.y, h = blockIdx.z;
  int bo = b * T_SEQ, t0 = qt * 32;
  int tid = threadIdx.x;

  // Stage Q, folding mask^2 * invscale into it (done once; bf16 in LDS).
#pragma unroll
  for (int it = 0; it < 8; ++it) {
    int idx = it * 2048 + tid * 8;
    int row = idx >> 9, col = idx & 511;
    v8bf qv = ld8(qbf + (size_t)(bo + t0 + row) * 512 + col);
    const float* mp = mq2 + h * 512 + col;
    v8bf o;
#pragma unroll
    for (int i = 0; i < 8; ++i) o[i] = (__bf16)((float)qv[i] * mp[i]);
    *(v8bf*)(&Qs[row][col]) = o;
  }
  if (tid < 32) { row_m[tid] = -3.0e38f; row_l[tid] = 0.f; }
  __syncthreads();

  int w = tid >> 5, lane = tid & 31, lhalf = lane >> 4, l15 = lane & 15;
  int mi = w & 1, di = w >> 1;  // wave owns ctx rows [mi*16,16) cols [di*128,128)
  v8f ctxacc[8] = {};
  int rr = tid >> 3, cc = tid & 7;

  for (int j = 0; j < 16; ++j) {
    int s0 = j * 64;
    // ---- S tile: Qm[16,512] @ K^T[512,16] (wave tile rows=mi*16, cols=di*16)
    v8f sc = {};
    const __bf16* Kp = kbf + (size_t)(bo + s0 + di * 16 + l15) * 512 + 16 * lhalf;
    const __bf16* Aq = &Qs[mi * 16 + l15][16 * lhalf];
    if (j + 1 < 16)
      __builtin_prefetch(kbf + (size_t)(bo + s0 + 64 + di * 16 + l15) * 512, 0, 3);
#pragma unroll
    for (int ks = 0; ks < 16; ++ks) {
      v16bf af  = mkfrag(ld8(Aq + ks * 32), ld8(Aq + ks * 32 + 8));
      v16bf bf_ = mkfrag(ld8(Kp + ks * 32), ld8(Kp + ks * 32 + 8));
      sc = wmma_bf16(af, bf_, sc);
    }
#pragma unroll
    for (int r = 0; r < 8; ++r)
      Ss[mi * 16 + r + 8 * lhalf][di * 16 + l15] = sc[r];
    __syncthreads();

    // ---- online softmax over the 64 fresh columns (32 rows x 8 threads) ----
    float sv[8], pm = -3.0e38f;
#pragma unroll
    for (int i = 0; i < 8; ++i) {
      int col = cc * 8 + i;
      float s = Ss[rr][col];
      if (kpm[b * T_SEQ + s0 + col]) s = -1.0e9f;
      sv[i] = s; pm = fmaxf(pm, s);
    }
    red[rr][cc] = pm;
    __syncthreads();
    if (cc == 0) {
      float mn = row_m[rr];
#pragma unroll
      for (int t = 0; t < 8; ++t) mn = fmaxf(mn, red[rr][t]);
      row_al[rr] = __expf(row_m[rr] - mn);
      row_m[rr] = mn;
    }
    __syncthreads();
    float mrow = row_m[rr], ps = 0.f;
#pragma unroll
    for (int i = 0; i < 8; ++i) {
      float p = __expf(sv[i] - mrow);
      Ps[rr][cc * 8 + i] = (__bf16)p;
      ps += p;
    }
    red[rr][cc] = ps;
    __syncthreads();
    if (cc == 0) {
      float t = 0.f;
#pragma unroll
      for (int k2 = 0; k2 < 8; ++k2) t += red[rr][k2];
      row_l[rr] = row_l[rr] * row_al[rr] + t;
    }
    __syncthreads();

    // ---- ctx = ctx * alpha + P @ V (V unmasked; mask applied at epilogue) --
    float al[8];
#pragma unroll
    for (int r = 0; r < 8; ++r) al[r] = row_al[mi * 16 + r + 8 * lhalf];
#pragma unroll
    for (int nt = 0; nt < 8; ++nt)
#pragma unroll
      for (int r = 0; r < 8; ++r) ctxacc[nt][r] *= al[r];
    const __bf16* Pp = &Ps[mi * 16 + l15][16 * lhalf];
#pragma unroll
    for (int ks = 0; ks < 2; ++ks) {
      v16bf pf = mkfrag(ld8(Pp + ks * 32), ld8(Pp + ks * 32 + 8));
#pragma unroll
      for (int nt = 0; nt < 8; ++nt) {
        int d = di * 128 + nt * 16 + l15;
        const __bf16* Vp = vtbf + (size_t)d * NTOK + bo + s0 + ks * 32 + 16 * lhalf;
        v16bf bv = mkfrag(ld8(Vp), ld8(Vp + 8));
        ctxacc[nt] = wmma_bf16(pf, bv, ctxacc[nt]);
      }
    }
    __syncthreads();
  }

  // ---- epilogue: divide by l, apply V-mask column scale, store per-head ----
  float li[8];
#pragma unroll
  for (int r = 0; r < 8; ++r) li[r] = 1.f / row_l[mi * 16 + r + 8 * lhalf];
#pragma unroll
  for (int nt = 0; nt < 8; ++nt) {
    int d = di * 128 + nt * 16 + l15;
    float mvv = mv[h * 512 + d];
#pragma unroll
    for (int r = 0; r < 8; ++r) {
      int Mg = bo + t0 + mi * 16 + r + 8 * lhalf;
      ctxh[(size_t)h * NTOK * 512 + (size_t)Mg * 512 + d] =
          (__bf16)(ctxacc[nt][r] * li[r] * mvv);
    }
  }
}

// ---------------- layernorm: one block per token row ------------------------
__global__ __launch_bounds__(256) void k_ln(const float* __restrict__ res,
                                            const float* __restrict__ gamma,
                                            const float* __restrict__ beta,
                                            float* __restrict__ out) {
  __shared__ float sb[256];
  int row = blockIdx.x, tid = threadIdx.x;
  const float* rp = res + (size_t)row * 512;
  float v0 = rp[tid], v1 = rp[tid + 256];
  sb[tid] = v0 + v1;
  __syncthreads();
  for (int s = 128; s > 0; s >>= 1) { if (tid < s) sb[tid] += sb[tid + s]; __syncthreads(); }
  float mu = sb[0] * (1.f / 512.f);
  __syncthreads();
  float d0 = v0 - mu, d1 = v1 - mu;
  sb[tid] = d0 * d0 + d1 * d1;
  __syncthreads();
  for (int s = 128; s > 0; s >>= 1) { if (tid < s) sb[tid] += sb[tid + s]; __syncthreads(); }
  float rs = rsqrtf(sb[0] * (1.f / 512.f) + 1e-5f);
  out[(size_t)row * 512 + tid]       = d0 * rs * gamma[tid]       + beta[tid];
  out[(size_t)row * 512 + tid + 256] = d1 * rs * gamma[tid + 256] + beta[tid + 256];
}

// ---------------------------------------------------------------------------
extern "C" void kernel_launch(void* const* d_in, const int* in_sizes, int n_in,
                              void* d_out, int out_size, void* d_ws, size_t ws_size,
                              hipStream_t stream) {
  const float* query = (const float*)d_in[0];
  const float* hw    = (const float*)d_in[1];
  const float* Wqkv  = (const float*)d_in[2];
  const float* bqkv  = (const float*)d_in[3];
  const float* Wout  = (const float*)d_in[4];
  const float* bout  = (const float*)d_in[5];
  const float* gamma = (const float*)d_in[6];
  const float* beta  = (const float*)d_in[7];
  const unsigned char* kpm = (const unsigned char*)d_in[8];
  float* out = (float*)d_out;

  char* ws = (char*)d_ws;
  size_t off = 0;
  auto take = [&](size_t bytes) { char* p = ws + off; off += (bytes + 255) & ~(size_t)255; return p; };
  float*  mq2    = (float*)take(NHEADS * D_MODEL * 4);
  float*  mv     = (float*)take(NHEADS * D_MODEL * 4);
  __bf16* xbf    = (__bf16*)take((size_t)NTOK * 512 * 2);
  __bf16* wqkvT  = (__bf16*)take((size_t)1536 * 512 * 2);
  __bf16* woutT  = (__bf16*)take((size_t)512 * 512 * 2);
  __bf16* qbf    = (__bf16*)take((size_t)NTOK * 512 * 2);
  __bf16* kbf    = (__bf16*)take((size_t)NTOK * 512 * 2);
  __bf16* vtbf   = (__bf16*)take((size_t)512 * NTOK * 2);
  __bf16* ctxh   = (__bf16*)take((size_t)NHEADS * NTOK * 512 * 2);
  __bf16* ctxbf  = (__bf16*)take((size_t)NTOK * 512 * 2);
  float*  res    = (float*)take((size_t)NTOK * 512 * 4);

  const int nED = NTOK * 512;  // 2,097,152

  k_prep<<<1, 512, 0, stream>>>(hw, mq2, mv);
  k_cvt_bf16<<<(nED + 255) / 256, 256, 0, stream>>>(query, xbf, nED);
  k_transpose_bf16<<<(512 * 1536 + 255) / 256, 256, 0, stream>>>(Wqkv, wqkvT, 512, 1536);
  k_transpose_bf16<<<(512 * 512 + 255) / 256, 256, 0, stream>>>(Wout, woutT, 512, 512);

  k_gemm<0><<<dim3(NTOK / 64, 1536 / 64), 256, 0, stream>>>(
      xbf, wqkvT, bqkv, qbf, kbf, vtbf, nullptr, nullptr);

  k_attn<<<dim3(T_SEQ / 32, 4, NHEADS), 256, 0, stream>>>(
      qbf, kbf, vtbf, mq2, mv, kpm, ctxh);

  k_headsum<<<(nED + 255) / 256, 256, 0, stream>>>(ctxh, ctxbf, nED);

  k_gemm<1><<<dim3(NTOK / 64, 512 / 64), 256, 0, stream>>>(
      ctxbf, woutT, bout, nullptr, nullptr, nullptr, query, res);

  k_ln<<<NTOK, 256, 0, stream>>>(res, gamma, beta, out);
  (void)in_sizes; (void)n_in; (void)out_size; (void)ws_size;
}
